// TransformerModule_42219528520059
// MI455X (gfx1250) — matrix operand order repaired and checked by reading, not compiled, not run on hardware
//
#include <hip/hip_runtime.h>
#include <hip/hip_bf16.h>

// ---------------------------------------------------------------------------
// Transformer attention block for MI455X (gfx1250), wave32 + WMMA bf16 + TDM.
// B=2, S=2048, D=2048, H=16, HD=128.
// Pipeline: pack (hidden->bf16, W->W^T bf16) | QKV GEMMs (TDM double-buffered
//           LDS staging + v_wmma_f32_16x16x32_bf16) | fused RMSNorm+RoPE |
//           flash attention (TDM staging, WMMA QK^T and PV) | output GEMM.
// ---------------------------------------------------------------------------

typedef __attribute__((ext_vector_type(16))) __bf16 v16bf;
typedef __attribute__((ext_vector_type(8)))  float  v8f;
typedef unsigned short u16;
typedef unsigned int   u32;
typedef unsigned long long u64;
typedef __attribute__((ext_vector_type(4))) unsigned int u32x4;
typedef __attribute__((ext_vector_type(8))) int i32x8;
typedef __attribute__((ext_vector_type(4))) int i32x4;

union AFrag { v16bf v; u32 u[8]; };
union CFrag { v8f  v; float f[8]; };

__device__ __forceinline__ u16 f2bf(float f) {
  u32 u = __builtin_bit_cast(u32, f);
  u32 r = u + 0x7FFFu + ((u >> 16) & 1u);   // round-to-nearest-even
  return (u16)(r >> 16);
}

// ---------------------------------------------------------------------------
// Tensor Data Mover: 2D tile load (bf16 elements) global -> LDS.
// D# layout per CDNA5 ISA 8.3/8.4: group0 = {count, lds_addr, global_addr,
// type=2}; group1 = {mask/data_size, tensor dims, tile dims, dim0 stride}.
// Issued by one wave; tracked with TENSORcnt.
// ---------------------------------------------------------------------------
__device__ __forceinline__ void tdm_load_2d_bf16(
    u32 lds_addr, const void* gptr,
    u32 tile_d0, u32 tile_d1,          // elements (d0 = contiguous)
    u32 tensor_d0, u32 tensor_d1,      // elements
    u64 stride_d0)                     // elements between dim1 lines
{
  u64 ga = (u64)(size_t)gptr;
  u32x4 g0;
  g0.x = 1u;                                               // count=1, user mode
  g0.y = lds_addr;                                         // LDS byte address
  g0.z = (u32)ga;                                          // global_addr[31:0]
  g0.w = ((u32)(ga >> 32) & 0x01FFFFFFu) | 0x80000000u;    // addr[56:32]|type=2
  i32x8 g1;
  g1[0] = (int)0x00010000u;                                // data_size=1 (2B)
  g1[1] = (int)((tensor_d0 & 0xFFFFu) << 16);              // tensor_dim0 lo
  g1[2] = (int)((tensor_d0 >> 16) | ((tensor_d1 & 0xFFFFu) << 16));
  g1[3] = (int)((tensor_d1 >> 16) | (tile_d0 << 16));      // tile_dim0
  g1[4] = (int)tile_d1;                                    // tile_dim1 (dim2=0)
  g1[5] = (int)(u32)(stride_d0 & 0xFFFFFFFFu);             // dim0 stride lo
  g1[6] = (int)(u32)((stride_d0 >> 32) & 0xFFFFu);         // dim0 stride hi
  g1[7] = 0;                                               // dim1 stride (2D)
  i32x4 z4 = {0, 0, 0, 0};
#if __clang_major__ >= 23
  i32x8 z8 = {0, 0, 0, 0, 0, 0, 0, 0};
  __builtin_amdgcn_tensor_load_to_lds(g0, g1, z4, z4, z8, 0);
#else
  __builtin_amdgcn_tensor_load_to_lds(g0, g1, z4, z4, 0);
#endif
}

__device__ __forceinline__ u32 lds_addr_of(const void* p) {
  return (u32)(size_t)p;   // flat LDS-aperture address: low 32 bits = LDS offset
}

// ---------------------------------------------------------------------------
// Elementwise f32 -> bf16 pack
// ---------------------------------------------------------------------------
__global__ void pack_bf16_kernel(const float* __restrict__ in,
                                 u16* __restrict__ out, long n) {
  long i = (long)blockIdx.x * blockDim.x + threadIdx.x;
  if (i < n) out[i] = f2bf(in[i]);
}

// ---------------------------------------------------------------------------
// W f32 [K][N] -> W^T bf16 [N][K], LDS-tiled 32x32 transpose.
// ---------------------------------------------------------------------------
__global__ __launch_bounds__(256) void packT_bf16_kernel(
    const float* __restrict__ in, u16* __restrict__ out, int K, int N) {
  __shared__ u16 t[32][33];
  const int tx = threadIdx.x & 31;
  const int ty = threadIdx.x >> 5;          // 0..7
  const int k0 = blockIdx.y * 32, n0 = blockIdx.x * 32;
#pragma unroll
  for (int i = 0; i < 4; ++i) {
    int k = ty + i * 8;
    t[tx][k] = f2bf(in[(size_t)(k0 + k) * N + n0 + tx]);
  }
  __syncthreads();
#pragma unroll
  for (int i = 0; i < 4; ++i) {
    int n = ty + i * 8;
    out[(size_t)(n0 + n) * K + k0 + tx] = t[n][tx];
  }
}

// ---------------------------------------------------------------------------
// V f32 [b,s,h,hd] -> bf16 [b,h,hd,s] (per-head transpose), LDS-tiled.
// ---------------------------------------------------------------------------
__global__ __launch_bounds__(256) void vtransT_kernel(
    const float* __restrict__ V, u16* __restrict__ out, int Sc, int Hc) {
  __shared__ u16 t[32][33];
  const int tx = threadIdx.x & 31;
  const int ty = threadIdx.x >> 5;
  const int s0 = blockIdx.x * 32;
  const int d0 = blockIdx.y * 32;
  const int bh = blockIdx.z;
  const int b = bh / Hc, h = bh % Hc;
#pragma unroll
  for (int i = 0; i < 4; ++i) {
    int s = ty + i * 8;
    t[tx][s] = f2bf(V[(((size_t)b * Sc + s0 + s) * Hc + h) * 128 + d0 + tx]);
  }
  __syncthreads();
#pragma unroll
  for (int i = 0; i < 4; ++i) {
    int d = ty + i * 8;
    out[((size_t)bh * 128 + d0 + d) * Sc + s0 + tx] = t[d][tx];
  }
}

// ---------------------------------------------------------------------------
// C[M,N] = A[M,K](bf16) @ Bt[N,K]^T(bf16) + bias[N], f32 out.
// BM=128 BN=128 BK=32, 256 threads (8 waves), each wave -> 32x64 (2x4 tiles).
// TDM double-buffered LDS staging; both tiles are plain [row][k] copies.
// ---------------------------------------------------------------------------
#define GBM 128
#define GBN 128
#define GBK 32

__global__ __launch_bounds__(256) void gemm_bf16_kernel(
    const u16* __restrict__ A, const u16* __restrict__ Bt,
    const float* __restrict__ bias, float* __restrict__ C,
    int M, int N, int K) {
  __shared__ u16 aS[2][GBM][GBK];   // [buf][m][k]
  __shared__ u16 bS[2][GBN][GBK];   // [buf][n][k]

  const int t    = threadIdx.x;
  const int lane = t & 31;
  const int wave = t >> 5;
  const int wm   = wave & 3;        // 0..3 along M
  const int wn   = wave >> 2;       // 0..1 along N
  const int half = lane >> 4;
  const int l16  = lane & 15;

  const int bm0 = blockIdx.y * GBM;
  const int bn0 = blockIdx.x * GBN;

  const u16* Ablk = A  + (size_t)bm0 * K;   // tile k-offset added per step
  const u16* Bblk = Bt + (size_t)bn0 * K;
  const u32 aBase = lds_addr_of(&aS[0][0][0]);
  const u32 bBase = lds_addr_of(&bS[0][0][0]);

  CFrag acc[2][4];
#pragma unroll
  for (int i = 0; i < 2; ++i)
#pragma unroll
    for (int j = 0; j < 4; ++j)
#pragma unroll
      for (int r = 0; r < 8; ++r) acc[i][j].f[r] = 0.0f;

  const int NK = K / GBK;
  if (wave == 0) {   // prologue: stage tile 0 into buffer 0
    tdm_load_2d_bf16(aBase, Ablk, GBK, GBM, (u32)K, (u32)M, (u64)K);
    tdm_load_2d_bf16(bBase, Bblk, GBK, GBN, (u32)K, (u32)N, (u64)K);
  }

  for (int kt = 0; kt < NK; ++kt) {
    const int buf = kt & 1, nxt = buf ^ 1;
    if (wave == 0) {
      if (kt + 1 < NK) {   // stage next tile, then wait for current (FIFO)
        const u16* an = Ablk + (size_t)(kt + 1) * GBK;
        const u16* bn = Bblk + (size_t)(kt + 1) * GBK;
        tdm_load_2d_bf16(aBase + nxt * (GBM * GBK * 2), an, GBK, GBM, (u32)K, (u32)M, (u64)K);
        tdm_load_2d_bf16(bBase + nxt * (GBN * GBK * 2), bn, GBK, GBN, (u32)K, (u32)N, (u64)K);
        __builtin_amdgcn_s_wait_tensorcnt((short)2);
      } else {
        __builtin_amdgcn_s_wait_tensorcnt((short)0);
      }
    }
    __syncthreads();   // current buffer ready for everyone

    AFrag af[2], bf[4];
#pragma unroll
    for (int mt = 0; mt < 2; ++mt) {
      int m = wm * 32 + mt * 16 + l16;
#pragma unroll
      for (int j = 0; j < 8; ++j) {
        int e = 2 * j;
        int k = e + half * 8 + (e >= 8 ? 8 : 0);
        af[mt].u[j] = *(const u32*)&aS[buf][m][k];
      }
    }
#pragma unroll
    for (int nt = 0; nt < 4; ++nt) {
      int n = wn * 64 + nt * 16 + l16;
#pragma unroll
      for (int j = 0; j < 8; ++j) {
        int k = 2 * j + half * 8 + (j >= 4 ? 8 : 0);
        bf[nt].u[j] = *(const u32*)&bS[buf][n][k];
      }
    }
#pragma unroll
    for (int mt = 0; mt < 2; ++mt)
#pragma unroll
      for (int nt = 0; nt < 4; ++nt)
        acc[mt][nt].v = __builtin_amdgcn_wmma_f32_16x16x32_bf16(
            false, af[mt].v, false, bf[nt].v, (short)0, acc[mt][nt].v,
            false, false);

    __syncthreads();   // all reads done before buffer is overwritten
  }

  // ---- epilogue: +bias, f32 store ----
#pragma unroll
  for (int mt = 0; mt < 2; ++mt) {
#pragma unroll
    for (int nt = 0; nt < 4; ++nt) {
      int n = bn0 + wn * 64 + nt * 16 + l16;
      float bv = bias[n];
#pragma unroll
      for (int r = 0; r < 8; ++r) {
        int m = bm0 + wm * 32 + mt * 16 + half * 8 + r;
        C[(size_t)m * N + n] = acc[mt][nt].f[r] + bv;
      }
    }
  }
}

// ---------------------------------------------------------------------------
// Fused RMSNorm (over HD=128) + RoPE; f32 [b,s,h,hd] -> bf16 [b,h,s,hd].
// One wave per head-row; lane owns pairs p=lane and p=lane+32.
// ---------------------------------------------------------------------------
__global__ __launch_bounds__(128) void normrope_kernel(
    const float* __restrict__ X, const float* __restrict__ w,
    const float* __restrict__ fcos, const float* __restrict__ fsin,
    u16* __restrict__ out, int Bc, int Sc, int Hc) {
  const int lane = threadIdx.x & 31;
  const int wave = threadIdx.x >> 5;
  long head = (long)blockIdx.x * 4 + wave;   // over B*S*H
  int h = (int)(head % Hc);
  long bs = head / Hc;
  int s = (int)(bs % Sc);
  int b = (int)(bs / Sc);

  const float* x = X + head * 128;
  float x0 = x[2 * lane],      x1 = x[2 * lane + 1];
  float x2 = x[2 * lane + 64], x3 = x[2 * lane + 65];
  float ss = x0 * x0 + x1 * x1 + x2 * x2 + x3 * x3;
#pragma unroll
  for (int m = 16; m >= 1; m >>= 1) ss += __shfl_xor(ss, m, 32);
  float inv = rsqrtf(ss * (1.0f / 128.0f) + 1e-6f);

  const float* fc = fcos + ((long)b * Sc + s) * 64;
  const float* fs = fsin + ((long)b * Sc + s) * 64;
  u16* o = out + (((long)b * Hc + h) * Sc + s) * 128;
  {
    float xr = x0 * inv * w[2 * lane], xi = x1 * inv * w[2 * lane + 1];
    float c = fc[lane], sn = fs[lane];
    float orr = xr * c - xi * sn, oi = xr * sn + xi * c;
    *(u32*)&o[2 * lane] = (u32)f2bf(orr) | ((u32)f2bf(oi) << 16);
  }
  {
    float xr = x2 * inv * w[2 * lane + 64], xi = x3 * inv * w[2 * lane + 65];
    float c = fc[lane + 32], sn = fs[lane + 32];
    float orr = xr * c - xi * sn, oi = xr * sn + xi * c;
    *(u32*)&o[2 * lane + 64] = (u32)f2bf(orr) | ((u32)f2bf(oi) << 16);
  }
}

// ---------------------------------------------------------------------------
// Flash attention. Grid (S/64, H, B); 128 threads = 4 waves; wave = 16 q-rows.
// Q/K bf16 [b,h,s,hd]; V bf16 [b,h,hd,s]; out bf16 [b,s,h,hd].
// TDM double-buffered K/V staging; WMMA for QK^T and PV; online softmax.
// ---------------------------------------------------------------------------
#define ATT_SCALE 0.08838834764831845f   // 1/sqrt(128)

__global__ __launch_bounds__(128) void attn_kernel(
    const u16* __restrict__ Qt, const u16* __restrict__ Kt,
    const u16* __restrict__ Vt, u16* __restrict__ Out, int Sc) {
  __shared__ u16 kS[2][32][128];    // [buf][key][hd]
  __shared__ u16 vS[2][128][32];    // [buf][hd][key]
  __shared__ u16 pS[4][16][32];     // per-wave P staging [row][key]

  const int lane = threadIdx.x & 31;
  const int wave = threadIdx.x >> 5;
  const int half = lane >> 4;
  const int l16  = lane & 15;
  const int b = blockIdx.z, h = blockIdx.y;
  const int Hc = gridDim.y;
  const int q0 = blockIdx.x * 64 + wave * 16;

  const u16* qbh = Qt + ((long)b * Hc + h) * (long)Sc * 128;
  const u16* kbh = Kt + ((long)b * Hc + h) * (long)Sc * 128;
  const u16* vbh = Vt + ((long)b * Hc + h) * (long)Sc * 128;  // [hd][s]
  const u32 kBase = lds_addr_of(&kS[0][0][0]);
  const u32 vBase = lds_addr_of(&vS[0][0][0]);

  // Q A-fragments for 4 hd-blocks of 32, kept in registers for the whole loop.
  AFrag qf[4];
#pragma unroll
  for (int d = 0; d < 4; ++d) {
    const u16* qrow = qbh + (long)(q0 + l16) * 128 + d * 32;
#pragma unroll
    for (int j = 0; j < 8; ++j) {
      int e = 2 * j;
      int k = e + half * 8 + (e >= 8 ? 8 : 0);
      qf[d].u[j] = *(const u32*)&qrow[k];
    }
  }

  CFrag o[8];
#pragma unroll
  for (int n = 0; n < 8; ++n)
#pragma unroll
    for (int r = 0; r < 8; ++r) o[n].f[r] = 0.0f;
  float mrow[8], lrow[8];
#pragma unroll
  for (int r = 0; r < 8; ++r) { mrow[r] = -1e30f; lrow[r] = 0.0f; }

  const int NB = Sc / 32;
  if (wave == 0) {   // prologue: stage key-block 0
    tdm_load_2d_bf16(kBase, kbh, 128, 32, 128u, (u32)Sc, 128ull);
    tdm_load_2d_bf16(vBase, vbh, 32, 128, (u32)Sc, 128u, (u64)Sc);
  }

  for (int kt = 0; kt < NB; ++kt) {
    const int buf = kt & 1, nxt = buf ^ 1;
    if (wave == 0) {
      if (kt + 1 < NB) {
        tdm_load_2d_bf16(kBase + nxt * 8192, kbh + (size_t)(kt + 1) * 32 * 128,
                         128, 32, 128u, (u32)Sc, 128ull);
        tdm_load_2d_bf16(vBase + nxt * 8192, vbh + (kt + 1) * 32,
                         32, 128, (u32)Sc, 128u, (u64)Sc);
        __builtin_amdgcn_s_wait_tensorcnt((short)2);
      } else {
        __builtin_amdgcn_s_wait_tensorcnt((short)0);
      }
    }
    __syncthreads();

    // ---- scores: two 16-key tiles, K-dim = HD via 4 chained WMMAs each ----
    CFrag s0, s1;
#pragma unroll
    for (int r = 0; r < 8; ++r) { s0.f[r] = 0.0f; s1.f[r] = 0.0f; }
#pragma unroll
    for (int d = 0; d < 4; ++d) {
      AFrag b0, b1;
#pragma unroll
      for (int j = 0; j < 8; ++j) {
        int k = d * 32 + 2 * j + half * 8 + (j >= 4 ? 8 : 0);          // hd index
        b0.u[j] = *(const u32*)&kS[buf][l16][k];
        b1.u[j] = *(const u32*)&kS[buf][16 + l16][k];
      }
      s0.v = __builtin_amdgcn_wmma_f32_16x16x32_bf16(false, qf[d].v, false, b0.v,
                                                     (short)0, s0.v, false, false);
      s1.v = __builtin_amdgcn_wmma_f32_16x16x32_bf16(false, qf[d].v, false, b1.v,
                                                     (short)0, s1.v, false, false);
    }

    // ---- online softmax (row stats via half-wave xor shuffles) ----
    float corr[8];
#pragma unroll
    for (int r = 0; r < 8; ++r) {
      float t0 = s0.f[r] * ATT_SCALE;
      float t1 = s1.f[r] * ATT_SCALE;
      float mx = fmaxf(t0, t1);
#pragma unroll
      for (int sh = 8; sh >= 1; sh >>= 1) mx = fmaxf(mx, __shfl_xor(mx, sh, 16));
      float mnew = fmaxf(mrow[r], mx);
      corr[r] = __expf(mrow[r] - mnew);
      mrow[r] = mnew;
      float p0 = __expf(t0 - mnew);
      float p1 = __expf(t1 - mnew);
      float ps = p0 + p1;
#pragma unroll
      for (int sh = 8; sh >= 1; sh >>= 1) ps += __shfl_xor(ps, sh, 16);
      lrow[r] = lrow[r] * corr[r] + ps;
      int row = half * 8 + r;
      pS[wave][row][l16]      = f2bf(p0);
      pS[wave][row][16 + l16] = f2bf(p1);
    }
#pragma unroll
    for (int n = 0; n < 8; ++n)
#pragma unroll
      for (int r = 0; r < 8; ++r) o[n].f[r] *= corr[r];

    // ---- P(16x32) @ V(32x128): 8 WMMAs ----
    AFrag pa;
#pragma unroll
    for (int j = 0; j < 8; ++j) {
      int e = 2 * j;
      int k = e + half * 8 + (e >= 8 ? 8 : 0);       // key index 0..31
      pa.u[j] = *(const u32*)&pS[wave][l16][k];
    }
#pragma unroll
    for (int n = 0; n < 8; ++n) {
      AFrag vb;
#pragma unroll
      for (int j = 0; j < 8; ++j) {
        int k = 2 * j + half * 8 + (j >= 4 ? 8 : 0);                   // key
        vb.u[j] = *(const u32*)&vS[buf][n * 16 + l16][k];
      }
      o[n].v = __builtin_amdgcn_wmma_f32_16x16x32_bf16(false, pa.v, false, vb.v,
                                                       (short)0, o[n].v, false, false);
    }
    __syncthreads();   // protect buffers before next TDM overwrite
  }

  // ---- finalize: /= rowsum, store bf16 [b,s,h,hd] ----
#pragma unroll
  for (int r = 0; r < 8; ++r) {
    float invl = 1.0f / lrow[r];
    int srow = q0 + half * 8 + r;
    u16* orow = Out + (((long)b * Sc + srow) * Hc + h) * 128;
#pragma unroll
    for (int n = 0; n < 8; ++n) orow[n * 16 + l16] = f2bf(o[n].f[r] * invl);
  }
}

// ---------------------------------------------------------------------------
// Host-side orchestration
// ---------------------------------------------------------------------------
extern "C" void kernel_launch(void* const* d_in, const int* in_sizes, int n_in,
                              void* d_out, int out_size, void* d_ws, size_t ws_size,
                              hipStream_t stream) {
  (void)in_sizes; (void)n_in; (void)out_size; (void)ws_size;
  const float* hs   = (const float*)d_in[0];
  const float* Wq   = (const float*)d_in[1];
  const float* bq   = (const float*)d_in[2];
  const float* Wk   = (const float*)d_in[3];
  const float* bk   = (const float*)d_in[4];
  const float* Wv   = (const float*)d_in[5];
  const float* bv   = (const float*)d_in[6];
  const float* Wo   = (const float*)d_in[7];
  const float* bo   = (const float*)d_in[8];
  const float* qnw  = (const float*)d_in[9];
  const float* knw  = (const float*)d_in[10];
  const float* fcos = (const float*)d_in[11];
  const float* fsin = (const float*)d_in[12];

  const int Bc = 2, Sc = 2048, Dc = 2048, Hc = 16;
  const long Mr = (long)Bc * Sc;            // 4096 rows
  const long nHS = Mr * Dc;                 // 8M elements
  const long nW  = (long)Dc * Dc;           // 4M elements

  char* p = (char*)d_ws;
  u16*   hs_b = (u16*)p;   p += nHS * 2;
  u16*   wq_b = (u16*)p;   p += nW * 2;     // transposed [N][K]
  u16*   wk_b = (u16*)p;   p += nW * 2;
  u16*   wv_b = (u16*)p;   p += nW * 2;
  u16*   wo_b = (u16*)p;   p += nW * 2;
  float* qf   = (float*)p; p += nHS * 4;
  float* kf   = (float*)p; p += nHS * 4;
  float* vf   = (float*)p; p += nHS * 4;
  u16*   qt   = (u16*)p;   p += nHS * 2;
  u16*   kt   = (u16*)p;   p += nHS * 2;
  u16*   vt   = (u16*)p;   p += nHS * 2;    // [b,h,hd,s]
  u16*   at_b = (u16*)p;   p += nHS * 2;

  const int thr = 256;
  // 1) pack hidden to bf16; weights to transposed bf16
  pack_bf16_kernel<<<(int)((nHS + thr - 1) / thr), thr, 0, stream>>>(hs, hs_b, nHS);
  dim3 gt(Dc / 32, Dc / 32);
  packT_bf16_kernel<<<gt, 256, 0, stream>>>(Wq, wq_b, Dc, Dc);
  packT_bf16_kernel<<<gt, 256, 0, stream>>>(Wk, wk_b, Dc, Dc);
  packT_bf16_kernel<<<gt, 256, 0, stream>>>(Wv, wv_b, Dc, Dc);
  packT_bf16_kernel<<<gt, 256, 0, stream>>>(Wo, wo_b, Dc, Dc);

  // 2) QKV projections
  dim3 gg(Dc / GBN, (int)(Mr / GBM));
  gemm_bf16_kernel<<<gg, 256, 0, stream>>>(hs_b, wq_b, bq, qf, (int)Mr, Dc, Dc);
  gemm_bf16_kernel<<<gg, 256, 0, stream>>>(hs_b, wk_b, bk, kf, (int)Mr, Dc, Dc);
  gemm_bf16_kernel<<<gg, 256, 0, stream>>>(hs_b, wv_b, bv, vf, (int)Mr, Dc, Dc);

  // 3) RMSNorm + RoPE (q,k); V per-head transpose
  int nrBlocks = (Bc * Sc * Hc) / 4;
  normrope_kernel<<<nrBlocks, 128, 0, stream>>>(qf, qnw, fcos, fsin, qt, Bc, Sc, Hc);
  normrope_kernel<<<nrBlocks, 128, 0, stream>>>(kf, knw, fcos, fsin, kt, Bc, Sc, Hc);
  dim3 gv(Sc / 32, 128 / 32, Bc * Hc);
  vtransT_kernel<<<gv, 256, 0, stream>>>(vf, vt, Sc, Hc);

  // 4) flash attention
  dim3 ga(Sc / 64, Hc, Bc);
  attn_kernel<<<ga, 128, 0, stream>>>(qt, kt, vt, at_b, Sc);

  // 5) output projection
  gemm_bf16_kernel<<<gg, 256, 0, stream>>>(at_b, wo_b, bo, (float*)d_out, (int)Mr, Dc, Dc);
}
